// P4DTransConv_68436008895045
// MI455X (gfx1250) — compile-verified, matrix operand order checked by reading, not run on hardware
//
#include <hip/hip_runtime.h>

// ---------- problem constants ----------
#define BB 8
#define TT 4
#define NN 1024
#define MM 4096
#define BT (BB*TT)
#define C_IN 256
#define C_ORG 128
#define K1 (C_IN + C_ORG)   // 384
#define P1 256
#define P2 256

typedef __attribute__((ext_vector_type(16))) __bf16 v16bf;
typedef __attribute__((ext_vector_type(8)))  __bf16 v8bf;
typedef __attribute__((ext_vector_type(8)))  float  v8f;
typedef __attribute__((ext_vector_type(4)))  float  v4f;

union BFrag { v16bf v; v8bf h[2]; };
union FAcc  { v8f v; v4f q[2]; };

// ---------------------------------------------------------------------------
// 1) three_nn: per query point, 3 nearest of 1024 reference points (LDS tile)
// ---------------------------------------------------------------------------
__global__ void three_nn_kernel(const float* __restrict__ xyzs,
                                const float* __restrict__ oxyz,
                                float* __restrict__ w3,
                                int* __restrict__ idx3) {
  __shared__ float sx[NN], sy[NN], sz[NN];
  const int bt = blockIdx.x;
  for (int j = threadIdx.x; j < NN; j += 256) {
    const float* p = xyzs + ((size_t)bt * NN + j) * 3;
    sx[j] = p[0]; sy[j] = p[1]; sz[j] = p[2];
  }
  __syncthreads();

  const int m = blockIdx.y * 256 + threadIdx.x;
  const float* q = oxyz + ((size_t)bt * MM + m) * 3;
  const float qx = q[0], qy = q[1], qz = q[2];

  float d0 = 1e30f, d1 = 1e30f, d2 = 1e30f;
  int   i0 = 0,     i1 = 0,     i2 = 0;
  for (int j = 0; j < NN; ++j) {
    const float dx = qx - sx[j], dy = qy - sy[j], dz = qz - sz[j];
    const float d = dx * dx + dy * dy + dz * dz;
    if (d < d0)      { d2 = d1; i2 = i1; d1 = d0; i1 = i0; d0 = d; i0 = j; }
    else if (d < d1) { d2 = d1; i2 = i1; d1 = d;  i1 = j; }
    else if (d < d2) { d2 = d;  i2 = j; }
  }
  const float w0 = 1.0f / (d0 + 1e-8f);
  const float w1 = 1.0f / (d1 + 1e-8f);
  const float w2 = 1.0f / (d2 + 1e-8f);
  const float inv = 1.0f / (w0 + w1 + w2);
  const size_t base = ((size_t)bt * MM + m) * 3;
  w3[base + 0] = w0 * inv; w3[base + 1] = w1 * inv; w3[base + 2] = w2 * inv;
  idx3[base + 0] = i0;     idx3[base + 1] = i1;     idx3[base + 2] = i2;
}

// ---------------------------------------------------------------------------
// 2) build X1 = concat(interp, orig_feats) in [bt][M][K1] bf16 (k contiguous)
// ---------------------------------------------------------------------------
__global__ void build_x1_kernel(const float* __restrict__ feats,
                                const float* __restrict__ ofeats,
                                const float* __restrict__ w3,
                                const int* __restrict__ idx3,
                                __bf16* __restrict__ X1) {
  const int bt = blockIdx.x;
  const int m  = blockIdx.y;
  const size_t pbase = ((size_t)bt * MM + m) * 3;
  const float w0 = w3[pbase + 0], w1 = w3[pbase + 1], w2 = w3[pbase + 2];
  const int   i0 = idx3[pbase + 0], i1 = idx3[pbase + 1], i2 = idx3[pbase + 2];

  const int c = threadIdx.x;                               // 0..255
  const float* f = feats + ((size_t)bt * C_IN + c) * NN;
  const float v = w0 * f[i0] + w1 * f[i1] + w2 * f[i2];
  __bf16* xr = X1 + ((size_t)bt * MM + m) * K1;
  xr[c] = (__bf16)v;
  if (c < C_ORG) {
    xr[C_IN + c] = (__bf16)ofeats[((size_t)bt * C_ORG + c) * MM + m];
  }
}

// ---------------------------------------------------------------------------
// 3) f32 -> bf16 conversion (weights)
// ---------------------------------------------------------------------------
__global__ void cvt_bf16_kernel(const float* __restrict__ src,
                                __bf16* __restrict__ dst, int n) {
  const int i = blockIdx.x * 256 + threadIdx.x;
  if (i < n) dst[i] = (__bf16)src[i];
}

// ---------------------------------------------------------------------------
// 4) WMMA GEMM: Y[bt][m][o] = sum_k W[o][k] * X[bt][m][k]
//    A = W bf16 [256][K] row-major, B = X bf16 [M][K] row-major.
//    One wave computes a 16(o) x 64(m) tile: 4 accumulators, A reused 4x.
//    Fragment layouts per CDNA5 ISA §7.12.2 (16-bit, wave32):
//      lanes 0-15: K 0..7 & 16..23,  lanes 16-31: K 8..15 & 24..31.
// ---------------------------------------------------------------------------
__global__ void wmma_gemm_kernel(const __bf16* __restrict__ Wb,
                                 const __bf16* __restrict__ X,
                                 float* __restrict__ Y,
                                 int K) {
  const int bt   = blockIdx.x;
  const int o0   = blockIdx.y * 16;
  const int lane = threadIdx.x & 31;
  const int wv   = threadIdx.x >> 5;                  // 8 waves / block
  const int m0   = (blockIdx.z * 8 + wv) * 64;

  X += (size_t)bt * MM * K;
  Y += (size_t)bt * MM * 256;

  const int half  = (lane < 16) ? 0 : 8;              // K sub-offset per lane
  const int row   = o0 + (lane & 15);                 // A: output-channel row
  const int mbase = m0 + (lane & 15);                 // B/D: column (per lane)
  const __bf16* Arow = Wb + (size_t)row * K;

  v8f acc0 = {}, acc1 = {}, acc2 = {}, acc3 = {};

  for (int k0 = 0; k0 < K; k0 += 32) {
    BFrag a;
    a.h[0] = *(const v8bf*)(Arow + k0 + half);
    a.h[1] = *(const v8bf*)(Arow + k0 + half + 16);

    BFrag b0, b1, b2, b3;
    const __bf16* x0 = X + (size_t)(mbase +  0) * K + k0 + half;
    const __bf16* x1 = X + (size_t)(mbase + 16) * K + k0 + half;
    const __bf16* x2 = X + (size_t)(mbase + 32) * K + k0 + half;
    const __bf16* x3 = X + (size_t)(mbase + 48) * K + k0 + half;
    b0.h[0] = *(const v8bf*)(x0); b0.h[1] = *(const v8bf*)(x0 + 16);
    b1.h[0] = *(const v8bf*)(x1); b1.h[1] = *(const v8bf*)(x1 + 16);
    b2.h[0] = *(const v8bf*)(x2); b2.h[1] = *(const v8bf*)(x2 + 16);
    b3.h[0] = *(const v8bf*)(x3); b3.h[1] = *(const v8bf*)(x3 + 16);

    acc0 = __builtin_amdgcn_wmma_f32_16x16x32_bf16(false, a.v, false, b0.v,
                                                   (short)0, acc0, false, false);
    acc1 = __builtin_amdgcn_wmma_f32_16x16x32_bf16(false, a.v, false, b1.v,
                                                   (short)0, acc1, false, false);
    acc2 = __builtin_amdgcn_wmma_f32_16x16x32_bf16(false, a.v, false, b2.v,
                                                   (short)0, acc2, false, false);
    acc3 = __builtin_amdgcn_wmma_f32_16x16x32_bf16(false, a.v, false, b3.v,
                                                   (short)0, acc3, false, false);
  }

  // D layout: VGPR r, lanes 0-15 -> (o0+r, mbase); lanes 16-31 -> (o0+8+r, mbase)
  const int obase = o0 + half;
  v8f accs[4] = {acc0, acc1, acc2, acc3};
#pragma unroll
  for (int s = 0; s < 4; ++s) {
    FAcc u; u.v = accs[s];
    float* dst = Y + (size_t)(mbase + 16 * s) * 256 + obase;
    *(v4f*)(dst + 0) = u.q[0];
    *(v4f*)(dst + 4) = u.q[1];
  }
}

// ---------------------------------------------------------------------------
// 5) BN stats per (t, o): deterministic tree reduction over (B, M)
// ---------------------------------------------------------------------------
__global__ void bn_stats_kernel(const float* __restrict__ Y,
                                const float* __restrict__ gamma,
                                const float* __restrict__ beta,
                                float* __restrict__ scale,
                                float* __restrict__ shift) {
  const int to = blockIdx.x;           // t*256 + o
  const int t = to >> 8, o = to & 255;
  float s1 = 0.0f, s2 = 0.0f;
  for (int j = threadIdx.x; j < BB * MM; j += 256) {
    const int b = j >> 12, m = j & (MM - 1);
    const float v = Y[((size_t)(b * TT + t) * MM + m) * 256 + o];
    s1 += v; s2 += v * v;
  }
  __shared__ float r1[256], r2[256];
  r1[threadIdx.x] = s1; r2[threadIdx.x] = s2;
  __syncthreads();
  for (int st = 128; st > 0; st >>= 1) {
    if (threadIdx.x < st) {
      r1[threadIdx.x] += r1[threadIdx.x + st];
      r2[threadIdx.x] += r2[threadIdx.x + st];
    }
    __syncthreads();
  }
  if (threadIdx.x == 0) {
    const float cnt  = (float)(BB * MM);
    const float mean = r1[0] / cnt;
    const float var  = r2[0] / cnt - mean * mean;
    const float sc   = gamma[o] * rsqrtf(var + 1e-5f);
    scale[to] = sc;
    shift[to] = beta[o] - mean * sc;
  }
}

// ---------------------------------------------------------------------------
// 6) BN + ReLU -> bf16 (input to GEMM2), layout preserved [bt][m][o]
// ---------------------------------------------------------------------------
__global__ void bn_relu_bf16_kernel(const float* __restrict__ Y,
                                    const float* __restrict__ scale,
                                    const float* __restrict__ shift,
                                    __bf16* __restrict__ X2) {
  const int row = blockIdx.x;          // bt*M + m
  const int t = (row >> 12) & (TT - 1);
  const int o = threadIdx.x;
  const float v = Y[(size_t)row * 256 + o];
  const float r = fmaxf(v * scale[t * 256 + o] + shift[t * 256 + o], 0.0f);
  X2[(size_t)row * 256 + o] = (__bf16)r;
}

// ---------------------------------------------------------------------------
// 7) BN + ReLU + transpose to output layout [bt][o][m] (LDS 16x16 tile)
// ---------------------------------------------------------------------------
__global__ void bn_relu_out_kernel(const float* __restrict__ Y,
                                   const float* __restrict__ scale,
                                   const float* __restrict__ shift,
                                   float* __restrict__ out) {
  __shared__ float tile[16][17];
  const int bt = blockIdx.z;
  const int t  = bt & (TT - 1);
  const int m0 = blockIdx.x * 16, o0 = blockIdx.y * 16;
  const int tx = threadIdx.x, ty = threadIdx.y;

  const int o = o0 + tx, m = m0 + ty;
  const float v = Y[((size_t)bt * MM + m) * 256 + o];
  tile[ty][tx] = fmaxf(v * scale[t * 256 + o] + shift[t * 256 + o], 0.0f);
  __syncthreads();
  out[((size_t)bt * 256 + (o0 + ty)) * MM + (m0 + tx)] = tile[tx][ty];
}

// ---------------------------------------------------------------------------
extern "C" void kernel_launch(void* const* d_in, const int* in_sizes, int n_in,
                              void* d_out, int out_size, void* d_ws, size_t ws_size,
                              hipStream_t stream) {
  const float* xyzs   = (const float*)d_in[0];
  const float* oxyz   = (const float*)d_in[1];
  const float* feats  = (const float*)d_in[2];
  const float* ofeats = (const float*)d_in[3];
  const float* W1     = (const float*)d_in[4];
  const float* g1     = (const float*)d_in[5];
  const float* b1     = (const float*)d_in[6];
  const float* W2     = (const float*)d_in[7];
  const float* g2     = (const float*)d_in[8];
  const float* b2     = (const float*)d_in[9];
  (void)in_sizes; (void)n_in; (void)out_size; (void)ws_size;

  // workspace partition (256B aligned)
  char* p = (char*)d_ws;
  auto take = [&](size_t bytes) {
    void* r = (void*)p;
    p += (bytes + 255) & ~(size_t)255;
    return r;
  };
  float*  w3     = (float*)take((size_t)BT * MM * 3 * sizeof(float));
  int*    idx3   = (int*)  take((size_t)BT * MM * 3 * sizeof(int));
  __bf16* W1b    = (__bf16*)take((size_t)P1 * K1 * sizeof(__bf16));
  __bf16* W2b    = (__bf16*)take((size_t)P2 * P1 * sizeof(__bf16));
  __bf16* X1     = (__bf16*)take((size_t)BT * MM * K1 * sizeof(__bf16));
  float*  Y1     = (float*)take((size_t)BT * MM * P1 * sizeof(float));
  __bf16* X2     = (__bf16*)take((size_t)BT * MM * P1 * sizeof(__bf16));
  float*  Y2     = (float*)take((size_t)BT * MM * P2 * sizeof(float));
  float*  scale1 = (float*)take((size_t)TT * P1 * sizeof(float));
  float*  shift1 = (float*)take((size_t)TT * P1 * sizeof(float));
  float*  scale2 = (float*)take((size_t)TT * P2 * sizeof(float));
  float*  shift2 = (float*)take((size_t)TT * P2 * sizeof(float));

  float* out = (float*)d_out;
  float* out_x = out + (size_t)BT * MM * 3;   // after original_xyzs passthrough

  // output tuple part 0: original_xyzs passthrough
  hipMemcpyAsync(out, oxyz, (size_t)BT * MM * 3 * sizeof(float),
                 hipMemcpyDeviceToDevice, stream);

  // weight conversion
  cvt_bf16_kernel<<<(P1 * K1 + 255) / 256, 256, 0, stream>>>(W1, W1b, P1 * K1);
  cvt_bf16_kernel<<<(P2 * P1 + 255) / 256, 256, 0, stream>>>(W2, W2b, P2 * P1);

  // three_nn: grid (bt, M/256)
  three_nn_kernel<<<dim3(BT, MM / 256), 256, 0, stream>>>(xyzs, oxyz, w3, idx3);

  // build X1: grid (bt, m)
  build_x1_kernel<<<dim3(BT, MM), 256, 0, stream>>>(feats, ofeats, w3, idx3, X1);

  // GEMM1: grid (bt, O/16, M/(8*64))
  wmma_gemm_kernel<<<dim3(BT, P1 / 16, MM / 512), 256, 0, stream>>>(W1b, X1, Y1, K1);
  bn_stats_kernel<<<TT * P1, 256, 0, stream>>>(Y1, g1, b1, scale1, shift1);
  bn_relu_bf16_kernel<<<BT * MM, 256, 0, stream>>>(Y1, scale1, shift1, X2);

  // GEMM2
  wmma_gemm_kernel<<<dim3(BT, P2 / 16, MM / 512), 256, 0, stream>>>(W2b, X2, Y2, P1);
  bn_stats_kernel<<<TT * P2, 256, 0, stream>>>(Y2, g2, b2, scale2, shift2);
  bn_relu_out_kernel<<<dim3(MM / 16, P2 / 16, BT), dim3(16, 16), 0, stream>>>(
      Y2, scale2, shift2, out_x);
}
